// HeteroGAT_38165079392696
// MI455X (gfx1250) — matrix operand order repaired and checked by reading, not compiled, not run on hardware
//
#include <hip/hip_runtime.h>

// ---------------- problem constants (match reference) ----------------
#define HGAT_H   8
#define HGAT_C   32
#define HID      256
#define NT_N     100000
#define NU_N     50000
#define NE_E     300000
#define IN_T     64
#define IN_U     32

typedef __attribute__((ext_vector_type(16))) _Float16 v16h;
typedef __attribute__((ext_vector_type(8)))  float    v8f;

// order-preserving float <-> uint encoding for atomicMax-based segment max
__device__ __forceinline__ unsigned fenc(float f) {
    unsigned u = __float_as_uint(f);
    return (u & 0x80000000u) ? ~u : (u | 0x80000000u);
}
__device__ __forceinline__ float fdec(unsigned u) {
    return (u & 0x80000000u) ? __uint_as_float(u & 0x7FFFFFFFu) : __uint_as_float(~u);
}

// ---- CDNA5 async global->LDS copy (16B per lane), ASYNCcnt-tracked ----
__device__ __forceinline__ void async_copy_b128(unsigned lds_off, const void* gaddr) {
    asm volatile("global_load_async_to_lds_b128 %0, %1, off"
                 :: "v"(lds_off), "v"(gaddr) : "memory");
}
__device__ __forceinline__ void wait_asynccnt0() {
    asm volatile("s_wait_asynccnt 0x0" ::: "memory");
}

// ---------------------------------------------------------------------
// Pack W[K x 256] (f32, row-major) into f16 B-fragments:
//   frag idx = (kb*16 + t)*32 + lane, 16 halves each.
//   Per lane: col = t*16 + (lane&15); K = kb*32 + ((lane<16)?0:16) + i.
// ---------------------------------------------------------------------
__global__ void k_packW(const float* __restrict__ W, _Float16* __restrict__ Wh, int K) {
    int idx = blockIdx.x * blockDim.x + threadIdx.x;
    int total = (K >> 5) * 16 * 32;
    if (idx >= total) return;
    int lane = idx & 31;
    int t    = (idx >> 5) & 15;
    int kb   = idx >> 9;
    int kbase = (kb << 5) + ((lane < 16) ? 0 : 16);
    int col   = t * 16 + (lane & 15);
    v16h h;
#pragma unroll
    for (int i = 0; i < 16; ++i) h[i] = (_Float16)W[(long)(kbase + i) * HID + col];
    *(v16h*)(Wh + (long)idx * 16) = h;
}

// ---------------------------------------------------------------------
// WMMA GEMM: Cout[N x 256] = A[N x K] * W[K x 256]  (f16 mul, f32 acc)
// act==1: Cout = elu(Cout + bias)
// Block = 256 thr = 8 waves = 4(M) x 2(N); wave tile 16 x 128 (8 acc tiles).
// B K-blocks are staged to LDS with async copies, double-buffered; A-frag
// float4 loads for kb+1 are issued before the kb WMMA burst.
// ---------------------------------------------------------------------
__global__ void k_gemm(const float* __restrict__ A, const _Float16* __restrict__ Wh,
                       const float* __restrict__ bias, float* __restrict__ Cout,
                       int N, int K, int act) {
    __shared__ __align__(64) _Float16 Bs[2][8192];   // 2 x 16KB staged B K-blocks

    const int tid   = threadIdx.x;
    const int lane  = tid & 31;
    const int wave  = tid >> 5;
    const int waveM = wave & 3;
    const int waveN = wave >> 2;                 // 0/1: which 128-col half
    const int row0  = blockIdx.x * 64 + waveM * 16;

    const int mrow = row0 + (lane & 15);
    const int arow = (mrow < N) ? mrow : (N - 1);   // clamp; stores are guarded
    const int kA   = (lane < 16) ? 0 : 16;          // A-frag K base within 32
    const int bcol = lane & 15;
    const int nkb  = K >> 5;

    // cooperative B staging: 16KB per K-block = 256 threads x 64B (4 x b128)
    const char* wsrc = (const char*)Wh;
    auto stageB = [&](int kb, int buf) {
        unsigned ldst = (unsigned)(size_t)&Bs[buf][0] + (unsigned)tid * 64u;
        const char* g = wsrc + (size_t)kb * 16384 + (size_t)tid * 64;
#pragma unroll
        for (int j = 0; j < 4; ++j)
            async_copy_b128(ldst + 16u * j, g + 16 * j);
    };

    v8f acc[8];
#pragma unroll
    for (int t = 0; t < 8; ++t) acc[t] = (v8f)0.0f;

    const float* arowp = A + (long)arow * K + kA;
    float4 q0 = *(const float4*)(arowp);
    float4 q1 = *(const float4*)(arowp + 4);
    float4 q2 = *(const float4*)(arowp + 8);
    float4 q3 = *(const float4*)(arowp + 12);

    stageB(0, 0);
    wait_asynccnt0();
    __syncthreads();

    for (int kb = 0; kb < nkb; ++kb) {
        const int cur = kb & 1;
        if (kb + 1 < nkb) stageB(kb + 1, cur ^ 1);  // async, overlaps compute

        // issue A loads for kb+1 early (latency hidden under WMMAs)
        float4 n0, n1, n2, n3;
        if (kb + 1 < nkb) {
            const float* np = arowp + (kb + 1) * 32;
            n0 = *(const float4*)(np);
            n1 = *(const float4*)(np + 4);
            n2 = *(const float4*)(np + 8);
            n3 = *(const float4*)(np + 12);
        }

        v16h af;
        af[0]  = (_Float16)q0.x; af[1]  = (_Float16)q0.y;
        af[2]  = (_Float16)q0.z; af[3]  = (_Float16)q0.w;
        af[4]  = (_Float16)q1.x; af[5]  = (_Float16)q1.y;
        af[6]  = (_Float16)q1.z; af[7]  = (_Float16)q1.w;
        af[8]  = (_Float16)q2.x; af[9]  = (_Float16)q2.y;
        af[10] = (_Float16)q2.z; af[11] = (_Float16)q2.w;
        af[12] = (_Float16)q3.x; af[13] = (_Float16)q3.y;
        af[14] = (_Float16)q3.z; af[15] = (_Float16)q3.w;

        // B-frags from LDS: frag idx = (waveN*8 + t)*32 + lane, 32B each
        const v16h* bvl = (const v16h*)&Bs[cur][(((waveN * 8) * 32 + lane)) * 16];
#pragma unroll
        for (int t = 0; t < 8; ++t) {
            v16h bf = bvl[t * 32];
            acc[t] = __builtin_amdgcn_wmma_f32_16x16x32_f16(
                false, af, false, bf, (short)0, acc[t], false, false);
        }

        q0 = n0; q1 = n1; q2 = n2; q3 = n3;

        if (kb + 1 < nkb) wait_asynccnt0();  // staged data visible before barrier
        __syncthreads();                     // protect buffer reuse
    }

    // ---- store: D layout — VGPR r holds row (r + 8*(lane>=16)), col = lane&15
    const int rofs = (lane < 16) ? 0 : 8;
#pragma unroll
    for (int t = 0; t < 8; ++t) {
        int col = (waveN * 8 + t) * 16 + bcol;
#pragma unroll
        for (int r = 0; r < 8; ++r) {
            int row = row0 + rofs + r;
            if (row < N) {
                float v = acc[t][r];
                if (act) { v += bias[col]; v = v > 0.f ? v : expm1f(v); }
                Cout[(long)row * HID + col] = v;
            }
        }
    }
}

// a[n,h] = sum_c X[n, h*32+c] * att[h*32+c]
__global__ void k_att(const float* __restrict__ X, const float* __restrict__ att,
                      float* __restrict__ out, int N) {
    int t = blockIdx.x * blockDim.x + threadIdx.x;
    if (t >= N * HGAT_H) return;
    int n = t >> 3, h = t & 7;
    const float4* x = (const float4*)(X + (long)n * HID + h * HGAT_C);
    const float4* a = (const float4*)(att + h * HGAT_C);
    float s = 0.f;
#pragma unroll
    for (int c = 0; c < 8; ++c) {
        float4 xv = x[c], av = a[c];
        s += xv.x * av.x + xv.y * av.y + xv.z * av.z + xv.w * av.w;
    }
    out[t] = s;
}

__global__ void k_fill_u32(unsigned* __restrict__ p, unsigned v, long n) {
    long i = blockIdx.x * (long)blockDim.x + threadIdx.x;
    if (i < n) p[i] = v;
}

// out[n, :] = b1[:] (+ b2[:])
__global__ void k_init_bias(float* __restrict__ out, const float* __restrict__ b1,
                            const float* __restrict__ b2, long n) {
    long i = blockIdx.x * (long)blockDim.x + threadIdx.x;
    if (i >= n) return;
    int col = (int)(i & (HID - 1));
    float v = b1[col];
    if (b2) v += b2[col];
    out[i] = v;
}

// pass 1: e = leaky_relu(a_src[src]+a_dst[dst], 0.2); segment max via atomicMax(enc)
__global__ void k_edge_logit(const int* __restrict__ src, const int* __restrict__ dst,
                             const float* __restrict__ as, const float* __restrict__ ad,
                             float* __restrict__ ebuf, unsigned* __restrict__ menc) {
    int t = blockIdx.x * blockDim.x + threadIdx.x;
    if (t >= NE_E * HGAT_H) return;
    int e = t >> 3, h = t & 7;
    int s = src[e], d = dst[e];
    float v = as[s * HGAT_H + h] + ad[d * HGAT_H + h];
    v = v > 0.f ? v : 0.2f * v;
    ebuf[t] = v;
    atomicMax(&menc[d * HGAT_H + h], fenc(v));
}

// pass 2: ee = exp(e - m[dst]); denom[dst] += ee
__global__ void k_edge_exp(const int* __restrict__ dst, float* __restrict__ ebuf,
                           const unsigned* __restrict__ menc, float* __restrict__ denom) {
    int t = blockIdx.x * blockDim.x + threadIdx.x;
    if (t >= NE_E * HGAT_H) return;
    int e = t >> 3, h = t & 7;
    int d = dst[e];
    float m  = fdec(menc[d * HGAT_H + h]);
    float ee = expf(ebuf[t] - m);
    ebuf[t]  = ee;
    atomicAdd(&denom[d * HGAT_H + h], ee);
}

// pass 3: one wave per edge; lane covers feature (k*32+lane), head k
__global__ void k_edge_scatter(const int* __restrict__ src, const int* __restrict__ dst,
                               const float* __restrict__ hs, const float* __restrict__ ebuf,
                               const float* __restrict__ denom, float* __restrict__ out) {
    int lane = threadIdx.x & 31, wave = threadIdx.x >> 5;
    int e = blockIdx.x * 8 + wave;
    if (e >= NE_E) return;
    int s = src[e], d = dst[e];
    const float* hp = hs + (long)s * HID;
    float*       op = out + (long)d * HID;
#pragma unroll
    for (int k = 0; k < HGAT_H; ++k) {
        float alpha = ebuf[e * HGAT_H + k] / (denom[d * HGAT_H + k] + 1e-16f);
        int f = k * 32 + lane;
        atomicAdd(&op[f], hp[f] * alpha);
    }
}

__global__ void k_elu(const float* __restrict__ in, float* __restrict__ out, long n) {
    long i = blockIdx.x * (long)blockDim.x + threadIdx.x;
    if (i < n) { float v = in[i]; out[i] = v > 0.f ? v : expm1f(v); }
}

// classifier: wave per node, 256-dot + shuffle reduce
__global__ void k_clf(const float* __restrict__ h, const float* __restrict__ Wc,
                      const float* __restrict__ bc, float* __restrict__ out, int N) {
    int lane = threadIdx.x & 31, wave = threadIdx.x >> 5;
    int n = blockIdx.x * 8 + wave;
    if (n >= N) return;
    const float* x = h + (long)n * HID;
    float s = 0.f;
#pragma unroll
    for (int j = 0; j < 8; ++j) s += x[lane + 32 * j] * Wc[lane + 32 * j];
#pragma unroll
    for (int off = 16; off > 0; off >>= 1) s += __shfl_xor(s, off, 32);
    if (lane == 0) out[n] = s + bc[0];
}

// ---------------------------------------------------------------------
extern "C" void kernel_launch(void* const* d_in, const int* in_sizes, int n_in,
                              void* d_out, int out_size, void* d_ws, size_t ws_size,
                              hipStream_t stream) {
    (void)in_sizes; (void)n_in; (void)out_size; (void)ws_size;
    // ---- inputs: insertion-order recursive flatten of setup_inputs() ----
    int ix = 0;
    const float* Wt = (const float*)d_in[ix++];
    const float* bt = (const float*)d_in[ix++];
    const float* Wu = (const float*)d_in[ix++];
    const float* bu = (const float*)d_in[ix++];
    struct GatP { const float *W, *as, *ad, *b; };
    GatP g[2][3];                       // [layer][0=ut,1=tu,2=tt]
    for (int l = 0; l < 2; ++l)
        for (int e = 0; e < 3; ++e) {
            g[l][e].W  = (const float*)d_in[ix++];
            g[l][e].as = (const float*)d_in[ix++];
            g[l][e].ad = (const float*)d_in[ix++];
            g[l][e].b  = (const float*)d_in[ix++];
        }
    const float* Wc  = (const float*)d_in[ix++];
    const float* bc  = (const float*)d_in[ix++];
    const float* x_t = (const float*)d_in[ix++];
    const float* x_u = (const float*)d_in[ix++];
    const int* src_ut = (const int*)d_in[ix++];
    const int* dst_ut = (const int*)d_in[ix++];
    const int* src_tu = (const int*)d_in[ix++];
    const int* dst_tu = (const int*)d_in[ix++];
    const int* src_tt = (const int*)d_in[ix++];
    const int* dst_tt = (const int*)d_in[ix++];

    // ---- workspace bump allocator ----
    char* wp = (char*)d_ws;
    auto alloc = [&](size_t bytes) -> void* {
        void* p = wp; wp += (bytes + 255) & ~(size_t)255; return p;
    };
    float*    h_t   = (float*)alloc((size_t)NT_N * HID * 4);
    float*    h_u   = (float*)alloc((size_t)NU_N * HID * 4);
    float*    out_t = (float*)alloc((size_t)NT_N * HID * 4);
    float*    out_u = (float*)alloc((size_t)NU_N * HID * 4);
    float*    buf   = (float*)alloc((size_t)NT_N * HID * 4);   // transformed src feats
    float*    a_src = (float*)alloc((size_t)NT_N * HGAT_H * 4);
    float*    a_dst = (float*)alloc((size_t)NT_N * HGAT_H * 4);
    unsigned* menc  = (unsigned*)alloc((size_t)NT_N * HGAT_H * 4);
    float*    denom = (float*)alloc((size_t)NT_N * HGAT_H * 4);
    float*    ebuf  = (float*)alloc((size_t)NE_E * HGAT_H * 4);
    _Float16* Wh_t  = (_Float16*)alloc((size_t)IN_T * HID * 2);
    _Float16* Wh_u  = (_Float16*)alloc((size_t)IN_U * HID * 2);
    _Float16* Wh_g[2][3];
    for (int l = 0; l < 2; ++l)
        for (int e = 0; e < 3; ++e)
            Wh_g[l][e] = (_Float16*)alloc((size_t)HID * HID * 2);

    auto packW = [&](const float* W, _Float16* Wh, int K) {
        int total = (K >> 5) * 16 * 32;
        k_packW<<<(total + 255) / 256, 256, 0, stream>>>(W, Wh, K);
    };
    packW(Wt, Wh_t, IN_T);
    packW(Wu, Wh_u, IN_U);
    for (int l = 0; l < 2; ++l)
        for (int e = 0; e < 3; ++e) packW(g[l][e].W, Wh_g[l][e], HID);

    auto gemm = [&](const float* A, const _Float16* Wh, const float* bias,
                    float* Cout, int N, int K, int act) {
        k_gemm<<<(N + 63) / 64, 256, 0, stream>>>(A, Wh, bias, Cout, N, K, act);
    };
    auto att = [&](const float* X, const float* a, float* o, int N) {
        k_att<<<(N * HGAT_H + 255) / 256, 256, 0, stream>>>(X, a, o, N);
    };
    auto run_gat = [&](const float* xs, int Ns, const float* xd, int Nd,
                       const GatP& p, const _Float16* Wh,
                       const int* src, const int* dst, float* out) {
        if (xs != xd) {                          // hd only needed for a_dst
            gemm(xd, Wh, nullptr, buf, Nd, HID, 0);
            att(buf, p.ad, a_dst, Nd);
        }
        gemm(xs, Wh, nullptr, buf, Ns, HID, 0);  // hs kept in buf for messages
        att(buf, p.as, a_src, Ns);
        if (xs == xd) att(buf, p.ad, a_dst, Nd);
        long nh = (long)Nd * HGAT_H;
        k_fill_u32<<<(nh + 255) / 256, 256, 0, stream>>>(menc, 0u, nh);
        k_fill_u32<<<(nh + 255) / 256, 256, 0, stream>>>((unsigned*)denom, 0u, nh);
        long eh = (long)NE_E * HGAT_H;
        k_edge_logit<<<(eh + 255) / 256, 256, 0, stream>>>(src, dst, a_src, a_dst, ebuf, menc);
        k_edge_exp  <<<(eh + 255) / 256, 256, 0, stream>>>(dst, ebuf, menc, denom);
        k_edge_scatter<<<(NE_E + 7) / 8, 256, 0, stream>>>(src, dst, buf, ebuf, denom, out);
    };

    // ---- input projections (fused bias + ELU) ----
    gemm(x_t, Wh_t, bt, h_t, NT_N, IN_T, 1);
    gemm(x_u, Wh_u, bu, h_u, NU_N, IN_U, 1);

    // ---- two hetero-GAT layers ----
    for (int l = 0; l < 2; ++l) {
        long nt = (long)NT_N * HID, nu = (long)NU_N * HID;
        k_init_bias<<<(nt + 255) / 256, 256, 0, stream>>>(out_t, g[l][0].b, g[l][2].b, nt);
        k_init_bias<<<(nu + 255) / 256, 256, 0, stream>>>(out_u, g[l][1].b, nullptr,   nu);
        run_gat(h_u, NU_N, h_t, NT_N, g[l][0], Wh_g[l][0], src_ut, dst_ut, out_t); // u->t
        run_gat(h_t, NT_N, h_t, NT_N, g[l][2], Wh_g[l][2], src_tt, dst_tt, out_t); // t->t
        run_gat(h_t, NT_N, h_u, NU_N, g[l][1], Wh_g[l][1], src_tu, dst_tu, out_u); // t->u
        k_elu<<<(nt + 255) / 256, 256, 0, stream>>>(out_t, h_t, nt);
        k_elu<<<(nu + 255) / 256, 256, 0, stream>>>(out_u, h_u, nu);
    }

    // ---- classifier ----
    k_clf<<<(NT_N + 7) / 8, 256, 0, stream>>>(h_t, Wc, bc, (float*)d_out, NT_N);
}
